// _xLSTMNetwork_23347442221180
// MI455X (gfx1250) — matrix operand order repaired and checked by reading, not compiled, not run on hardware
//
#include <hip/hip_runtime.h>
#include <hip/hip_bf16.h>

// ---------------- problem constants ----------------
#define B_  4
#define S_  1024
#define F_  64
#define D_  512
#define L_  4
#define NH_ 4
#define I_  1024          // 2*D
#define DH_ 256           // I/NH
#define PB_ 4
#define PH_ 256           // I/PB
#define KS_ 4
#define NC_ 10
#define ROWS_ (B_*S_)     // 4096

typedef __attribute__((ext_vector_type(16))) _Float16 v16h;
typedef __attribute__((ext_vector_type(8)))  _Float16 v8h;
typedef __attribute__((ext_vector_type(8)))  float    v8f;

__device__ __forceinline__ v16h cat8(v8h lo, v8h up) {
  return __builtin_shufflevector(lo, up, 0,1,2,3,4,5,6,7,8,9,10,11,12,13,14,15);
}
// A-matrix 16x32 f16 fragment; p already includes lane-hi offset (+8 halves for lanes>=16)
__device__ __forceinline__ v16h load_a_frag(const _Float16* p) {
  return cat8(*(const v8h*)p, *(const v8h*)(p + 16));
}
// B-matrix 32x16 f16 fragment; p already includes lane-hi offset (+8 halves for lanes>=16)
__device__ __forceinline__ v16h load_b_frag(const _Float16* p) {
  return cat8(*(const v8h*)p, *(const v8h*)(p + 16));
}

__device__ __forceinline__ float sigmoidf_(float x) { return 1.0f / (1.0f + __expf(-x)); }
__device__ __forceinline__ float siluf_(float x)    { return x * sigmoidf_(x); }

// ---------------- elementwise f32 -> f16 ----------------
__global__ void cvt_f16_kernel(const float* __restrict__ in, _Float16* __restrict__ out, int n) {
  int i = blockIdx.x * blockDim.x + threadIdx.x;
  if (i < n) out[i] = (_Float16)in[i];
}

// ---------------- f32 (KxN) -> f16 transposed (NxK) ----------------
__global__ void transpose_f16_kernel(const float* __restrict__ in, _Float16* __restrict__ out,
                                     int K, int N) {
  int idx = blockIdx.x * blockDim.x + threadIdx.x;
  if (idx >= K * N) return;
  const int k = idx % K, n = idx / K;
  out[(size_t)n * K + k] = (_Float16)in[(size_t)k * N + n];
}

// ---------------- WMMA GEMM: C[MxN] = A[MxK] @ B[KxN] (+bias)(+Cadd) -------------
// K compile-time: all fragment addresses = one base VGPR pair + constant immediate.
// BT is B transposed (N x K) f16. grid = (N/128, M/32), block = 32 (one wave).
template <int K>
__global__ void wmma_gemm_f16t_kernel(const _Float16* __restrict__ A,
                                      const _Float16* __restrict__ BT,
                                      const float* __restrict__ bias,
                                      const float* __restrict__ Cadd,
                                      float* __restrict__ C,
                                      int M, int N) {
  const int lane = threadIdx.x;
  const int l15 = lane & 15;
  const int m0 = blockIdx.y * 32;
  const int n0 = blockIdx.x * 128;
  v8f acc0[8] = {}, acc1[8] = {};
  const _Float16* pa = A + (size_t)(m0 + l15) * K + ((lane & 16) ? 8 : 0);
  const _Float16* pb = BT + (size_t)(n0 + l15) * K + ((lane & 16) ? 8 : 0);
  for (int k0 = 0; k0 < K; k0 += 32) {
    if (k0 + 32 < K) __builtin_prefetch(pa + 32, 0, 3);
    const v16h a0 = cat8(*(const v8h*)pa, *(const v8h*)(pa + 16));
    const v16h a1 = cat8(*(const v8h*)(pa + 16 * K), *(const v8h*)(pa + 16 * K + 16));
#pragma unroll
    for (int g = 0; g < 2; ++g) {
      v16h bfr[4];
#pragma unroll
      for (int i = 0; i < 4; ++i) {
        const _Float16* q = pb + (size_t)(g * 4 + i) * 16 * K;
        bfr[i] = cat8(*(const v8h*)q, *(const v8h*)(q + 16));
      }
#pragma unroll
      for (int i = 0; i < 4; ++i) {
        acc0[g * 4 + i] = __builtin_amdgcn_wmma_f32_16x16x32_f16(false, a0, false, bfr[i], (short)0, acc0[g * 4 + i], false, false);
        acc1[g * 4 + i] = __builtin_amdgcn_wmma_f32_16x16x32_f16(false, a1, false, bfr[i], (short)0, acc1[g * 4 + i], false, false);
      }
    }
    pa += 32; pb += 32;
  }
#pragma unroll
  for (int t = 0; t < 8; ++t) {
#pragma unroll
    for (int r = 0; r < 8; ++r) {
      const int cc = n0 + t * 16 + l15;
      const int rr0 = m0 + r + ((lane & 16) ? 8 : 0);
      const int rr1 = rr0 + 16;
      float v0 = acc0[t][r], v1 = acc1[t][r];
      if (bias) { v0 += bias[cc]; v1 += bias[cc]; }
      if (Cadd) { v0 += Cadd[(size_t)rr0 * N + cc]; v1 += Cadd[(size_t)rr1 * N + cc]; }
      C[(size_t)rr0 * N + cc] = v0;
      C[(size_t)rr1 * N + cc] = v1;
    }
  }
}

// ---------------- layernorm row (f32 in) -> f16 out ----------------
__global__ void ln_f16_kernel(const float* __restrict__ x, const float* __restrict__ w,
                              _Float16* __restrict__ out, int cols) {
  __shared__ float rs[256], rq[256];
  const int row = blockIdx.x, tid = threadIdx.x;
  const float* xr = x + (size_t)row * cols;
  float s = 0.f, q = 0.f;
  for (int c = tid; c < cols; c += 256) { float v = xr[c]; s += v; q += v * v; }
  rs[tid] = s; rq[tid] = q; __syncthreads();
  for (int st = 128; st > 0; st >>= 1) {
    if (tid < st) { rs[tid] += rs[tid + st]; rq[tid] += rq[tid + st]; }
    __syncthreads();
  }
  const float mean = rs[0] / cols;
  const float var = rq[0] / cols - mean * mean;
  const float rstd = rsqrtf(var + 1e-5f);
  for (int c = tid; c < cols; c += 256)
    out[(size_t)row * cols + c] = (_Float16)((xr[c] - mean) * rstd * w[c]);
}

// ---------------- depthwise causal conv (KS=4) + SiLU ----------------
__global__ void conv_silu_kernel(const float* __restrict__ up, const float* __restrict__ w,
                                 const float* __restrict__ bias, float* __restrict__ xc) {
  int idx = blockIdx.x * blockDim.x + threadIdx.x;
  if (idx >= B_ * S_ * I_) return;
  const int c = idx % I_;
  const int s = (idx / I_) % S_;
  const int b = idx / (I_ * S_);
  float acc = bias[c];
#pragma unroll
  for (int j = 0; j < KS_; ++j) {
    const int sp = s - (KS_ - 1) + j;
    if (sp >= 0) acc += up[((size_t)(b * S_ + sp)) * (2 * I_) + c] * w[c * KS_ + j];
  }
  xc[idx] = siluf_(acc);
}

// ---------------- q/k/v block-diagonal 4x4 + input/forget gates ----------------
// q16/k16 stored (B,NH,S,DH); vT stored transposed (B,NH,DH,S).
__global__ void qkv_gates_kernel(const float* __restrict__ up, const float* __restrict__ xc,
                                 const float* __restrict__ Wq, const float* __restrict__ Wk,
                                 const float* __restrict__ Wv,
                                 const float* __restrict__ Wi, const float* __restrict__ bi,
                                 const float* __restrict__ Wf, const float* __restrict__ bf,
                                 _Float16* __restrict__ q16, _Float16* __restrict__ k16,
                                 _Float16* __restrict__ vT,
                                 float* __restrict__ ig, float* __restrict__ fg) {
  __shared__ float red[8][256];
  const int bs = blockIdx.x;
  const int b = bs / S_, s = bs % S_;
  const int h = threadIdx.x;           // PH group
  const float* xcr = xc + (size_t)bs * I_;
  const float* xmr = up + (size_t)bs * (2 * I_);
  float xc4[4], xm4[4];
#pragma unroll
  for (int d = 0; d < 4; ++d) { xc4[d] = xcr[h * 4 + d]; xm4[d] = xmr[h * 4 + d]; }
  float pi[4] = {}, pf[4] = {};
#pragma unroll
  for (int o = 0; o < 4; ++o) {
    float qv = 0.f, kv = 0.f, vv = 0.f;
#pragma unroll
    for (int d = 0; d < 4; ++d) {
      qv += xc4[d] * Wq[h * 16 + o * 4 + d];
      kv += xc4[d] * Wk[h * 16 + o * 4 + d];
      vv += xm4[d] * Wv[h * 16 + o * 4 + d];
    }
    const int chan = h * 4 + o;
    const int n = chan >> 8, dd = chan & 255;
    const size_t qi = (((size_t)b * NH_ + n) * S_ + s) * DH_ + dd;
    q16[qi] = (_Float16)qv; k16[qi] = (_Float16)kv;
    vT[(((size_t)b * NH_ + n) * DH_ + dd) * S_ + s] = (_Float16)vv;
#pragma unroll
    for (int nh = 0; nh < 4; ++nh) {
      pi[nh] += qv * Wi[chan * NH_ + nh] + kv * Wi[(I_ + chan) * NH_ + nh] + vv * Wi[(2 * I_ + chan) * NH_ + nh];
      pf[nh] += qv * Wf[chan * NH_ + nh] + kv * Wf[(I_ + chan) * NH_ + nh] + vv * Wf[(2 * I_ + chan) * NH_ + nh];
    }
  }
#pragma unroll
  for (int nh = 0; nh < 4; ++nh) { red[nh][h] = pi[nh]; red[4 + nh][h] = pf[nh]; }
  __syncthreads();
  for (int st = 128; st > 0; st >>= 1) {
    if (h < st)
#pragma unroll
      for (int g = 0; g < 8; ++g) red[g][h] += red[g][h + st];
    __syncthreads();
  }
  if (h < 4)            ig[(((size_t)b * NH_ + h) * S_) + s]        = red[h][0] + bi[h];
  else if (h < 8)       fg[(((size_t)b * NH_ + (h - 4)) * S_) + s]  = red[h][0] + bf[h - 4];
}

// ---------------- per-head gate scan ----------------
__global__ void gate_scan_kernel(const float* __restrict__ ig, const float* __restrict__ fg,
                                 float* __restrict__ aarr, float* __restrict__ marr,
                                 float* __restrict__ nmarr) {
  const int hd = threadIdx.x;
  if (hd >= B_ * NH_) return;
  const float* igp = ig + (size_t)hd * S_;
  const float* fgp = fg + (size_t)hd * S_;
  float* ap = aarr + (size_t)hd * S_;
  float* mp = marr + (size_t)hd * S_;
  float* np = nmarr + (size_t)hd * S_;
  float cum = 0.f, rm = -1e30f;
  for (int s = 0; s < S_; ++s) {
    const float x = fgp[s];
    const float lf = fminf(x, 0.f) - log1pf(__expf(-fabsf(x)));   // stable log_sigmoid
    cum += lf;
    const float av = igp[s] - cum;
    rm = fmaxf(rm, av);
    ap[s] = av; mp[s] = rm; np[s] = cum + rm;
  }
}

// ---------------- mlstm streaming attention ----------------
// grid = (S/16, B*NH), block = 64 (2 waves). Wave w computes score subtile w (16 t-cols)
// and owns output columns [w*128, w*128+128) of DH. P + rowsums shared through LDS.
__global__ void __launch_bounds__(64)
mlstm_attn_kernel(const _Float16* __restrict__ q16, const _Float16* __restrict__ k16,
                  const _Float16* __restrict__ vT,
                  const float* __restrict__ aarr, const float* __restrict__ marr,
                  const float* __restrict__ nmarr, float* __restrict__ hout) {
  __shared__ _Float16 Pst[16][40];   // row stride 80B -> 16B-aligned chunks
  __shared__ float rsum2[2][16];
  const int tid = threadIdx.x;
  const int wid = tid >> 5;
  const int lane = tid & 31;
  const int l15 = lane & 15;
  const int hi8 = (lane & 16) ? 8 : 0;
  const int head = blockIdx.y;
  const int s0 = blockIdx.x * 16;
  const int tmax = s0 + 15;
  const _Float16* Q = q16 + (size_t)head * S_ * DH_;
  const _Float16* K = k16 + (size_t)head * S_ * DH_;
  const _Float16* Vt = vT + (size_t)head * DH_ * S_;
  const float* aa = aarr + (size_t)head * S_;
  const float* mm = marr + (size_t)head * S_;
  const float* nn = nmarr + (size_t)head * S_;
  const float scale = 0.0625f;   // 1/sqrt(DH)
  const int dbase = wid * 128;   // this wave's DH half

  // preload Q A-fragments (8 chunks of 32 along DH) -- both waves
  v16h qf[8];
  const _Float16* qrowp = Q + (size_t)(s0 + l15) * DH_ + hi8;
#pragma unroll
  for (int c = 0; c < 8; ++c) qf[c] = load_a_frag(qrowp + c * 32);

  float mrow[8];
#pragma unroll
  for (int r = 0; r < 8; ++r) mrow[r] = mm[s0 + r + hi8];

  v8f acc[8] = {};
  float rowsum[8] = {};

  for (int t0 = 0; t0 <= tmax; t0 += 32) {
    // ---- this wave's 16-col score subtile of QK^T, rescaled by exp(a[t]-m[s]) ----
    const int tb = t0 + wid * 16;
    v8f sc = {};
    if (tb <= tmax) {
      const _Float16* krowp = K + (size_t)(tb + l15) * DH_ + ((lane & 16) ? 8 : 0);
#pragma unroll
      for (int g = 0; g < 2; ++g) {
        v16h kf[4];
#pragma unroll
        for (int i = 0; i < 4; ++i) kf[i] = load_b_frag(krowp + (g * 4 + i) * 32);
#pragma unroll
        for (int i = 0; i < 4; ++i)
          sc = __builtin_amdgcn_wmma_f32_16x16x32_f16(false, qf[g * 4 + i], false, kf[i], (short)0, sc, false, false);
      }
    }
    const int tcol = tb + l15;
    const float av = (tb <= tmax) ? aa[tcol] : 0.f;
#pragma unroll
    for (int r = 0; r < 8; ++r) {
      const int srow = s0 + r + hi8;
      float val = 0.f;
      if (tb <= tmax && tcol <= srow)
        val = sc[r] * scale * __expf(av - mrow[r]);
      rowsum[r] += val;
      Pst[r + hi8][wid * 16 + l15] = (_Float16)val;
    }
    __syncthreads();
    // P as A-fragment (full 32 t-cols)
    v16h pf;
    {
      const _Float16* prow = &Pst[l15][hi8];
      pf = cat8(*(const v8h*)prow, *(const v8h*)(prow + 16));
    }
    // ---- acc += P @ V chunk over this wave's 128 output cols ----
#pragma unroll
    for (int dg = 0; dg < 2; ++dg) {
      v16h vf[4];
#pragma unroll
      for (int i = 0; i < 4; ++i) {
        const _Float16* vrow = Vt + (size_t)(dbase + (dg * 4 + i) * 16 + l15) * S_ + t0 + ((lane & 16) ? 8 : 0);
        vf[i] = load_b_frag(vrow);
      }
#pragma unroll
      for (int i = 0; i < 4; ++i)
        acc[dg * 4 + i] = __builtin_amdgcn_wmma_f32_16x16x32_f16(false, pf, false, vf[i], (short)0, acc[dg * 4 + i], false, false);
    }
    __syncthreads();   // protect Pst before next iteration's writes
  }

  // reduce this wave's partial rowsum across its 16 lanes, combine waves via LDS
#pragma unroll
  for (int r = 0; r < 8; ++r) {
    float v = rowsum[r];
    for (int off = 1; off < 16; off <<= 1) v += __shfl_xor(v, off, 32);
    rowsum[r] = v;
  }
  if (l15 == 0) {
#pragma unroll
    for (int r = 0; r < 8; ++r) rsum2[wid][r + hi8] = rowsum[r];
  }
  __syncthreads();
  float inv[8];
#pragma unroll
  for (int r = 0; r < 8; ++r) {
    const int row = r + hi8;
    const float total = rsum2[0][row] + rsum2[1][row];
    const float norm = fmaxf(fabsf(total), __expf(-nn[s0 + row]));
    inv[r] = 1.0f / (norm + 1e-6f);
  }
#pragma unroll
  for (int dt = 0; dt < 8; ++dt)
#pragma unroll
    for (int r = 0; r < 8; ++r) {
      const int srow = s0 + r + hi8;
      const int dcol = dbase + dt * 16 + l15;
      hout[((size_t)head * S_ + srow) * DH_ + dcol] = acc[dt][r] * inv[r];
    }
}

// ---------------- multi-head norm + skip + silu(z) -> y16 ----------------
__global__ void mhn_mix_kernel(const float* __restrict__ hatt, const float* __restrict__ up,
                               const float* __restrict__ xc, const float* __restrict__ mhn_w,
                               const float* __restrict__ skip, _Float16* __restrict__ y16) {
  __shared__ float rs[256], rq[256];
  const int id = blockIdx.x;
  const int s = id % S_;
  const int n = (id / S_) % NH_;
  const int b = id / (S_ * NH_);
  const int d = threadIdx.x;
  const float v = hatt[(((size_t)b * NH_ + n) * S_ + s) * DH_ + d];
  rs[d] = v; rq[d] = v * v; __syncthreads();
  for (int st = 128; st > 0; st >>= 1) {
    if (d < st) { rs[d] += rs[d + st]; rq[d] += rq[d + st]; }
    __syncthreads();
  }
  const float mean = rs[0] / DH_;
  const float var = rq[0] / DH_ - mean * mean;
  const float hn = (v - mean) * rsqrtf(var + 1e-5f);
  const int c = n * DH_ + d;
  const size_t bs = (size_t)b * S_ + s;
  const float z = up[bs * (2 * I_) + I_ + c];
  const float out = (hn * mhn_w[c] + skip[c] * xc[bs * I_ + c]) * siluf_(z);
  y16[bs * I_ + c] = (_Float16)out;
}

// ---------------- final: post-LN last token, MLP head ----------------
__global__ void head_kernel(const float* __restrict__ h, const float* __restrict__ post_ln_w,
                            const float* __restrict__ W1, const float* __restrict__ b1,
                            const float* __restrict__ W2, const float* __restrict__ b2,
                            float* __restrict__ out) {
  __shared__ float xs[D_];
  __shared__ float rs[256], rq[256];
  __shared__ float t1[D_ / 2];
  const int b = blockIdx.x, tid = threadIdx.x;
  const float* row = h + ((size_t)b * S_ + (S_ - 1)) * D_;
  float s = 0.f, q = 0.f;
  for (int c = tid; c < D_; c += 256) { float v = row[c]; s += v; q += v * v; }
  rs[tid] = s; rq[tid] = q; __syncthreads();
  for (int st = 128; st > 0; st >>= 1) {
    if (tid < st) { rs[tid] += rs[tid + st]; rq[tid] += rq[tid + st]; }
    __syncthreads();
  }
  const float mean = rs[0] / D_;
  const float var = rq[0] / D_ - mean * mean;
  const float rstd = rsqrtf(var + 1e-5f);
  for (int c = tid; c < D_; c += 256) xs[c] = (row[c] - mean) * rstd * post_ln_w[c];
  __syncthreads();
  {
    float acc = b1[tid];
    for (int k = 0; k < D_; ++k) acc += xs[k] * W1[k * (D_ / 2) + tid];
    t1[tid] = fmaxf(acc, 0.f);
  }
  __syncthreads();
  if (tid < NC_) {
    float acc = b2[tid];
    for (int k = 0; k < D_ / 2; ++k) acc += t1[k] * W2[k * NC_ + tid];
    out[b * NC_ + tid] = acc;
  }
}

// ---------------- host orchestration ----------------
extern "C" void kernel_launch(void* const* d_in, const int* in_sizes, int n_in,
                              void* d_out, int out_size, void* d_ws, size_t ws_size,
                              hipStream_t stream) {
  const float* x        = (const float*)d_in[0];
  const float* Wproj    = (const float*)d_in[1];
  const float* bproj    = (const float*)d_in[2];
  const float* ln_w     = (const float*)d_in[3];
  const float* Wup      = (const float*)d_in[4];
  const float* conv_w   = (const float*)d_in[5];
  const float* conv_b   = (const float*)d_in[6];
  const float* Wq       = (const float*)d_in[7];
  const float* Wk       = (const float*)d_in[8];
  const float* Wv       = (const float*)d_in[9];
  const float* Wi       = (const float*)d_in[10];
  const float* bi       = (const float*)d_in[11];
  const float* Wf       = (const float*)d_in[12];
  const float* bf       = (const float*)d_in[13];
  const float* mhn_w    = (const float*)d_in[14];
  const float* skip     = (const float*)d_in[15];
  const float* Wdown    = (const float*)d_in[16];
  const float* post_ln_w= (const float*)d_in[17];
  const float* W1       = (const float*)d_in[18];
  const float* b1       = (const float*)d_in[19];
  const float* W2       = (const float*)d_in[20];
  const float* b2       = (const float*)d_in[21];
  float* out = (float*)d_out;

  // workspace layout
  char* ws = (char*)d_ws;
  size_t off = 0;
  auto alloc = [&](size_t bytes) { char* p = ws + off; off += (bytes + 255) & ~size_t(255); return p; };
  float*    h        = (float*)   alloc((size_t)ROWS_ * D_ * 4);
  float*    up       = (float*)   alloc((size_t)ROWS_ * 2 * I_ * 4);
  float*    xc       = (float*)   alloc((size_t)ROWS_ * I_ * 4);
  float*    hatt     = (float*)   alloc((size_t)B_ * NH_ * S_ * DH_ * 4);
  _Float16* x16      = (_Float16*)alloc((size_t)ROWS_ * F_ * 2);
  _Float16* wprojT16 = (_Float16*)alloc((size_t)D_ * F_ * 2);
  _Float16* wupT16   = (_Float16*)alloc((size_t)L_ * 2 * I_ * D_ * 2);
  _Float16* wdownT16 = (_Float16*)alloc((size_t)L_ * D_ * I_ * 2);
  _Float16* xn16     = (_Float16*)alloc((size_t)ROWS_ * D_ * 2);
  _Float16* q16      = (_Float16*)alloc((size_t)ROWS_ * I_ * 2);
  _Float16* k16      = (_Float16*)alloc((size_t)ROWS_ * I_ * 2);
  _Float16* vT16     = (_Float16*)alloc((size_t)ROWS_ * I_ * 2);
  _Float16* y16      = (_Float16*)alloc((size_t)ROWS_ * I_ * 2);
  float*    ig       = (float*)   alloc((size_t)B_ * NH_ * S_ * 4);
  float*    fg       = (float*)   alloc((size_t)B_ * NH_ * S_ * 4);
  float*    aarr     = (float*)   alloc((size_t)B_ * NH_ * S_ * 4);
  float*    marr     = (float*)   alloc((size_t)B_ * NH_ * S_ * 4);
  float*    nmarr    = (float*)   alloc((size_t)B_ * NH_ * S_ * 4);
  (void)ws_size; (void)in_sizes; (void)n_in; (void)out_size;

  cvt_f16_kernel<<<(ROWS_ * F_ + 255) / 256, 256, 0, stream>>>(x, x16, ROWS_ * F_);
  // weights -> f16 transposed (N x K)
  transpose_f16_kernel<<<(F_ * D_ + 255) / 256, 256, 0, stream>>>(Wproj, wprojT16, F_, D_);
  for (int l = 0; l < L_; ++l) {
    transpose_f16_kernel<<<(D_ * 2 * I_ + 255) / 256, 256, 0, stream>>>(
        Wup + (size_t)l * D_ * 2 * I_, wupT16 + (size_t)l * 2 * I_ * D_, D_, 2 * I_);
    transpose_f16_kernel<<<(I_ * D_ + 255) / 256, 256, 0, stream>>>(
        Wdown + (size_t)l * I_ * D_, wdownT16 + (size_t)l * D_ * I_, I_, D_);
  }

  // h = x @ Wproj + bproj   (4096 x 64 x 512)
  wmma_gemm_f16t_kernel<F_><<<dim3(D_ / 128, ROWS_ / 32), 32, 0, stream>>>(
      x16, wprojT16, bproj, nullptr, h, ROWS_, D_);

  for (int l = 0; l < L_; ++l) {
    ln_f16_kernel<<<ROWS_, 256, 0, stream>>>(h, ln_w + (size_t)l * D_, xn16, D_);
    // up = xn @ Wup  (4096 x 512 x 2048)
    wmma_gemm_f16t_kernel<D_><<<dim3((2 * I_) / 128, ROWS_ / 32), 32, 0, stream>>>(
        xn16, wupT16 + (size_t)l * 2 * I_ * D_, nullptr, nullptr, up, ROWS_, 2 * I_);
    conv_silu_kernel<<<(B_ * S_ * I_ + 255) / 256, 256, 0, stream>>>(
        up, conv_w + (size_t)l * I_ * KS_, conv_b + (size_t)l * I_, xc);
    qkv_gates_kernel<<<ROWS_, 256, 0, stream>>>(
        up, xc,
        Wq + (size_t)l * PH_ * PB_ * PB_, Wk + (size_t)l * PH_ * PB_ * PB_,
        Wv + (size_t)l * PH_ * PB_ * PB_,
        Wi + (size_t)l * 3 * I_ * NH_, bi + (size_t)l * NH_,
        Wf + (size_t)l * 3 * I_ * NH_, bf + (size_t)l * NH_,
        q16, k16, vT16, ig, fg);
    gate_scan_kernel<<<1, 32, 0, stream>>>(ig, fg, aarr, marr, nmarr);
    mlstm_attn_kernel<<<dim3(S_ / 16, B_ * NH_), 64, 0, stream>>>(
        q16, k16, vT16, aarr, marr, nmarr, hatt);
    mhn_mix_kernel<<<B_ * NH_ * S_, 256, 0, stream>>>(
        hatt, up, xc, mhn_w + (size_t)l * I_, skip + (size_t)l * I_, y16);
    // h = h + y @ Wdown  (4096 x 1024 x 512)
    wmma_gemm_f16t_kernel<I_><<<dim3(D_ / 128, ROWS_ / 32), 32, 0, stream>>>(
        y16, wdownT16 + (size_t)l * D_ * I_, nullptr, h, h, ROWS_, D_);
  }

  head_kernel<<<B_, 256, 0, stream>>>(h, post_ln_w, W1, b1, W2, b2, out);
}
